// GPT2PagedAttention_16441134809762
// MI455X (gfx1250) — matrix operand order repaired and checked by reading, not compiled, not run on hardware
//
#include <hip/hip_runtime.h>

// ===========================================================================
// GPT-2 paged attention block on MI455X (gfx1250), bf16 WMMA pipeline.
//   K0 k_kvconv: f32 KV pages -> bf16 (K concat row-major, V transposed)
//   K1 k_qkv   : (8192x2048)@(2048x6144)+b -> q,k bf16 [B][H][Q][D], v [B][H][D][Q]
//                register-double-buffered staging
//   K2 k_attn  : flash attention; KV tiles ping-pong double-buffered in LDS via
//                global_load_async_to_lds_b128, s_wait_asynccnt 0x8 pipeline
//   K3 k_proj  : (8192x2048 bf16)@(2048x2048)+b -> f32 out, reg-double-buffered
// All matmuls: v_wmma_f32_16x16x32_bf16, f32 accumulate.
// ===========================================================================

typedef __bf16 bf16;
typedef __attribute__((ext_vector_type(16))) __bf16 v16bf;
typedef __attribute__((ext_vector_type(8)))  __bf16 v8bf;
typedef __attribute__((ext_vector_type(8)))  float  v8f;

#define WMMA_B16(a, b, c) \
  __builtin_amdgcn_wmma_f32_16x16x32_bf16(false, (a), false, (b), (short)0, (c), false, false)

// A fragment, 16-bit 16x32 (ISA 7.12.2): lane<16 holds K {kb..kb+7, kb+16..kb+23},
// kb=0; lane>=16 kb=8. rowbase points at (row, k_chunk_base) in LDS/global.
__device__ __forceinline__ v16bf fragA(const bf16* rowbase, int kb) {
  union { v16bf v; v8bf h[2]; } u;
  u.h[0] = *(const v8bf*)(rowbase + kb);
  u.h[1] = *(const v8bf*)(rowbase + kb + 16);
  return u.v;
}
// B fragment, 16-bit 32x16: lane half h holds K {8h..8h+7, 16+8h..16+8h+7},
// column = lane%16; colbase points at (col, k_chunk_base); kh = half*16.
__device__ __forceinline__ v16bf fragB(const bf16* colbase, int kh) {
  union { v16bf v; v8bf h[2]; } u;
  u.h[0] = *(const v8bf*)(colbase + (kh >> 1));
  u.h[1] = *(const v8bf*)(colbase + (kh >> 1) + 16);
  return u.v;
}
__device__ __forceinline__ v8f vzero8() {
  v8f z;
  for (int i = 0; i < 8; ++i) z[i] = 0.f;
  return z;
}

// Async global->LDS 16B copy per lane (GLOBAL_LOAD_ASYNC_TO_LDS_B128, GV mode,
// ASYNCcnt). lds_off = byte offset within block LDS alloc (low 32 bits of the
// flat shared address per the LDS aperture mapping).
__device__ __forceinline__ void async_g2l_b128(unsigned lds_off, const void* gptr) {
  asm volatile("global_load_async_to_lds_b128 %0, %1, off"
               :: "v"(lds_off), "v"(gptr) : "memory");
}
__device__ __forceinline__ unsigned lds_off32(const void* p) {
  return (unsigned)(size_t)p;
}

// ---------------------------------------------------------------------------
// K0: KV page conversion. kcat[b][h][page*256+p][d] = bf16(k_pages[...])
//     vt  [b][h][d][page*256+p] = bf16(v_pages[...])   (transposed via LDS)
// ---------------------------------------------------------------------------
__global__ __launch_bounds__(256)
void k_kvconv(const float* __restrict__ k_pages, const float* __restrict__ v_pages,
              bf16* __restrict__ kcat, bf16* __restrict__ vt) {
  const int page = blockIdx.x, b = blockIdx.y, h = blockIdx.z;
  const int tid = threadIdx.x;
  __shared__ bf16 lV[128 * 72];
  const size_t srcbase = ((((size_t)page * 8 + b) * 16 + h) * 256) * 128;
  const float* ksrc = k_pages + srcbase;
  const float* vsrc = v_pages + srcbase;
  bf16* kdst = kcat + (((size_t)b * 16 + h) * 2048 + page * 256) * 128;
  bf16* vdst = vt + ((size_t)b * 16 + h) * 128 * 2048 + page * 256;

  // K: straight copy/convert 256x128
  for (int i = tid; i < 8192; i += 256) {
    int r = i >> 5, c4 = (i & 31) << 2;
    float4 f = *(const float4*)(ksrc + r * 128 + c4);
    bf16* d = kdst + (size_t)r * 128 + c4;
    d[0] = (bf16)f.x; d[1] = (bf16)f.y; d[2] = (bf16)f.z; d[3] = (bf16)f.w;
  }
  // V: 4 tiles of 64(kv)x128(d), transpose via LDS, coalesced writes
  for (int t = 0; t < 4; ++t) {
    __syncthreads();
    for (int i = tid; i < 2048; i += 256) {
      int r = i >> 5, c4 = (i & 31) << 2;
      float4 g = *(const float4*)(vsrc + (size_t)(t * 64 + r) * 128 + c4);
      lV[(c4 + 0) * 72 + r] = (bf16)g.x;
      lV[(c4 + 1) * 72 + r] = (bf16)g.y;
      lV[(c4 + 2) * 72 + r] = (bf16)g.z;
      lV[(c4 + 3) * 72 + r] = (bf16)g.w;
    }
    __syncthreads();
    for (int i = tid; i < 1024; i += 256) {
      int r = i >> 3, c8 = (i & 7) << 3;
      *(v8bf*)(vdst + (size_t)r * 2048 + t * 64 + c8) = *(const v8bf*)&lV[r * 72 + c8];
    }
  }
}

// ---------------------------------------------------------------------------
// K1: QKV GEMM + bias -> q (pre-scaled), k row-major, v transposed, all bf16.
// Register double-buffered staging: global loads for step k+32 overlap WMMAs.
// ---------------------------------------------------------------------------
__global__ __launch_bounds__(256)
void k_qkv(const float* __restrict__ A, const float* __restrict__ W,
           const float* __restrict__ bias,
           bf16* __restrict__ qws, bf16* __restrict__ kws, bf16* __restrict__ vws_t) {
  const int K = 2048, N = 6144;
  __shared__ bf16 lA[128 * 40];   // [m][k], stride 40 (80B, 16B aligned, bank-spread)
  __shared__ bf16 lB[128 * 40];   // [n][k] (transposed on stage)
  const int tid  = threadIdx.x;
  const int lane = tid & 31, wave = tid >> 5;
  const int half = lane >> 4, l16 = lane & 15;
  const int kb = half * 8, kh = half * 16;
  const int m0 = blockIdx.x * 128;
  const int n0 = blockIdx.y * 128;
  const int wm = (wave & 3) * 32;
  const int wn = (wave >> 2) * 64;

  v8f acc[2][4];
  for (int i = 0; i < 2; ++i)
    for (int j = 0; j < 4; ++j) acc[i][j] = vzero8();

  float4 ga[4], gb[4];
  auto load_regs = [&](int k0) {
    for (int u = 0; u < 4; ++u) {
      const int ia = tid + u * 256;
      ga[u] = *(const float4*)(A + (size_t)(m0 + (ia >> 3)) * K + k0 + ((ia & 7) << 2));
      gb[u] = *(const float4*)(W + (size_t)(k0 + (ia >> 5)) * N + n0 + ((ia & 31) << 2));
    }
  };

  load_regs(0);
  for (int k0 = 0; k0 < K; k0 += 32) {
    __syncthreads();                          // previous compute done with LDS
    for (int u = 0; u < 4; ++u) {
      const int ia = tid + u * 256;
      bf16* d = &lA[(ia >> 3) * 40 + ((ia & 7) << 2)];
      d[0] = (bf16)ga[u].x; d[1] = (bf16)ga[u].y;
      d[2] = (bf16)ga[u].z; d[3] = (bf16)ga[u].w;
      const int r = ia >> 5, c4 = (ia & 31) << 2;
      lB[(c4 + 0) * 40 + r] = (bf16)gb[u].x;
      lB[(c4 + 1) * 40 + r] = (bf16)gb[u].y;
      lB[(c4 + 2) * 40 + r] = (bf16)gb[u].z;
      lB[(c4 + 3) * 40 + r] = (bf16)gb[u].w;
    }
    __syncthreads();                          // staging visible
    if (k0 + 32 < K) load_regs(k0 + 32);      // overlap next fetch with WMMAs
    v16bf af[2], bfr[4];
    for (int i = 0; i < 2; ++i) af[i] = fragA(&lA[(wm + i * 16 + l16) * 40], kb);
    for (int j = 0; j < 4; ++j) bfr[j] = fragB(&lB[(wn + j * 16 + l16) * 40], kh);
    for (int i = 0; i < 2; ++i)
      for (int j = 0; j < 4; ++j)
        acc[i][j] = WMMA_B16(af[i], bfr[j], acc[i][j]);
  }

  const float qscale = 0.08838834764831845f;  // 1/sqrt(128)
  for (int i = 0; i < 2; ++i)
    for (int j = 0; j < 4; ++j) {
      const int col = n0 + wn + j * 16 + l16;
      // col>>11 is uniform across the 16-wide tile: scalar -> uniform branch
      const int whichs = __builtin_amdgcn_readfirstlane(col >> 11);  // 0=q 1=k 2=v
      const int hh = (col & 2047) >> 7;
      const int dd = col & 127;
      const float bv = bias[col];
      for (int r = 0; r < 8; ++r) {
        const int row = m0 + wm + i * 16 + r + half * 8;
        const int bb = row >> 10, qq = row & 1023;
        const float v = acc[i][j][r] + bv;
        if (whichs == 0) {
          qws[((((size_t)bb * 16 + hh) * 1024) + qq) * 128 + dd] = (bf16)(v * qscale);
        } else if (whichs == 1) {
          kws[((((size_t)bb * 16 + hh) * 1024) + qq) * 128 + dd] = (bf16)v;
        } else {  // V stored transposed: [b][h][d][q]
          vws_t[((((size_t)bb * 16 + hh) * 128) + dd) * 1024 + qq] = (bf16)v;
        }
      }
    }
}

// ---------------------------------------------------------------------------
// K2: flash attention. Block = (q-tile of 128, head, batch); each of 8 waves
// owns 16 Q rows. KV tiles (64 wide) ping-pong between two LDS buffers:
// async-copy tile i+1 while WMMAs consume tile i; s_wait_asynccnt 0x8 retires
// exactly the older tile (async loads complete in order, 8 instrs per tile).
// Dynamic LDS: 2*lK(17408B) + 2*lV(18432B) + lP(18432B) = 90112B.
// Buffer pointers are computed with integer offsets (no pointer-array init of
// the extern shared symbol -> avoids addrspacecast-in-static-initializer).
// ---------------------------------------------------------------------------
__global__ __launch_bounds__(256)
void k_attn(const bf16* __restrict__ qws, const bf16* __restrict__ kws,
            const bf16* __restrict__ vws_t,
            const bf16* __restrict__ kcat, const bf16* __restrict__ vt,
            const int* __restrict__ page_pos, const int* __restrict__ input_pos,
            bf16* __restrict__ ctx) {
  extern __shared__ __align__(16) bf16 smem[];
  // element offsets within smem: lK[p] at p*8704, lV[p] at 17408 + p*9216,
  // lP at 35840 (8 waves * 16 rows * 72 stride)
  const int tid  = threadIdx.x;
  const int lane = tid & 31, wave = tid >> 5;
  const int half = lane >> 4, l16 = lane & 15;
  const int kb = half * 8, kh = half * 16;
  const int qt = blockIdx.x, h = blockIdx.y, b = blockIdx.z;
  const int qbase = qt * 128 + wave * 16;

  // Q A-fragments: rows qbase..+15, d_head = 128 -> 4 K-chunks of 32
  const bf16* qrow = qws + ((((size_t)b * 16 + h) * 1024) + qbase + l16) * 128;
  v16bf qf[4];
  for (int c = 0; c < 4; ++c) qf[c] = fragA(qrow + c * 32, kb);

  float mr[8], lr[8];
  v8f acc[8];
  for (int r = 0; r < 8; ++r) { mr[r] = -3.0e38f; lr[r] = 0.f; }
  for (int t = 0; t < 8; ++t) acc[t] = vzero8();

  const int len_cur = input_pos[b * 2 + 1];
  bf16* pw = smem + 35840 + wave * 16 * 72;

  const int ncur = qt * 2 + 2;         // causal bound: kv tile start <= max row
  const int ntiles = 32 + ncur;        // 2048/64 paged tiles + current tiles

  // issue tile ti's async copies into buffer p (8 async instrs per wave)
  auto issue = [&](int ti, int p) {
    const bf16* ks;
    const bf16* vs;
    int vstride;
    if (ti < 32) {
      ks = kcat + (((size_t)b * 16 + h) * 2048 + ti * 64) * 128;
      vs = vt + ((size_t)b * 16 + h) * 128 * 2048 + ti * 64;
      vstride = 2048;
    } else {
      const int t = ti - 32;
      ks = kws + ((((size_t)b * 16 + h) * 1024) + t * 64) * 128;
      vs = vws_t + ((size_t)b * 16 + h) * 128 * 1024 + t * 64;
      vstride = 1024;
    }
    const unsigned lKo = lds_off32(smem) + (unsigned)p * 17408u;           // bytes
    const unsigned lVo = lds_off32(smem) + 34816u + (unsigned)p * 18432u;  // bytes
    for (int i = tid; i < 1024; i += 256) {
      int r = i >> 4, c8 = (i & 15) << 3;
      async_g2l_b128(lKo + (unsigned)(r * 272 + c8 * 2), ks + (size_t)r * 128 + c8);
    }
    for (int i = tid; i < 1024; i += 256) {
      int r = i >> 3, c8 = (i & 7) << 3;
      async_g2l_b128(lVo + (unsigned)(r * 144 + c8 * 2), vs + (size_t)r * vstride + c8);
    }
  };

  issue(0, 0);
  for (int ti = 0; ti < ntiles; ++ti) {
    const int p = ti & 1;
    if (ti + 1 < ntiles) {
      issue(ti + 1, p ^ 1);                       // buf p^1 freed by barrier below
      asm volatile("s_wait_asynccnt 0x8" ::: "memory");  // tile ti landed
    } else {
      asm volatile("s_wait_asynccnt 0x0" ::: "memory");
    }
    __syncthreads();                              // all waves' tile ti visible
    const bf16* lK = smem + p * 8704;
    const bf16* lV = smem + 17408 + p * 9216;

    int mlo, mhi, kvoff;
    bool causal;
    if (ti < 32) {
      const int page = ti >> 2;
      const int ps = page_pos[(page * 8 + b) * 2 + 0];
      const int pl = page_pos[(page * 8 + b) * 2 + 1];
      mlo = page * 256 + ps;
      mhi = page * 256 + ps + pl;
      kvoff = ti * 64;
      causal = false;
    } else {
      mlo = 0; mhi = len_cur;
      kvoff = (ti - 32) * 64;
      causal = true;
    }

    // ---- S = (Q*scale) K^T : 4 column tiles x 4 K-chunks ----
    v8f s[4];
    for (int j = 0; j < 4; ++j) {
      v8f sa = vzero8();
      for (int c = 0; c < 4; ++c) {
        v16bf bk = fragB(&lK[(j * 16 + l16) * 136 + c * 32], kh);
        sa = WMMA_B16(qf[c], bk, sa);
      }
      s[j] = sa;
    }

    // ---- mask + online softmax (rows live across 16-lane halves) ----
    for (int r = 0; r < 8; ++r) {
      const int qrow_abs = qbase + r + half * 8;
      float mx = -3.0e38f;
      for (int j = 0; j < 4; ++j) {
        const int kv = kvoff + j * 16 + l16;
        bool ok = (kv >= mlo) && (kv < mhi);
        if (causal) ok = ok && (kv <= qrow_abs);
        const float v = ok ? s[j][r] : -3.0e38f;
        s[j][r] = v;
        if (v > mx) mx = v;
      }
      for (int d = 1; d < 16; d <<= 1) {
        float o = __shfl_xor(mx, d, 32);
        if (o > mx) mx = o;
      }
      const float mn = mr[r] > mx ? mr[r] : mx;
      const float al = __expf(mr[r] - mn);
      mr[r] = mn;
      lr[r] *= al;
      for (int t = 0; t < 8; ++t) acc[t][r] *= al;
      float sum = 0.f;
      for (int j = 0; j < 4; ++j) {
        const float e = __expf(s[j][r] - mn);
        s[j][r] = e;
        sum += e;
      }
      for (int d = 1; d < 16; d <<= 1) sum += __shfl_xor(sum, d, 32);
      lr[r] += sum;
    }

    // ---- P -> per-wave LDS (C layout store, A layout gather) ----
    for (int j = 0; j < 4; ++j)
      for (int r = 0; r < 8; ++r)
        pw[(r + half * 8) * 72 + j * 16 + l16] = (bf16)s[j][r];
    asm volatile("s_wait_dscnt 0x0" ::: "memory");  // intra-wave LDS RAW

    // ---- O += P V : 8 d-tiles x 2 K-chunks over kv ----
    for (int kc = 0; kc < 2; ++kc) {
      v16bf pf = fragA(&pw[l16 * 72 + kc * 32], kb);
      for (int nd = 0; nd < 8; ++nd) {
        v16bf vf = fragB(&lV[(nd * 16 + l16) * 72 + kc * 32], kh);
        acc[nd] = WMMA_B16(pf, vf, acc[nd]);
      }
    }
    __syncthreads();   // all waves done reading buf p -> safe for issue(ti+2)
  }

  // epilogue: ctx[b][q][h*128+d] = acc / max(l, 1e-9)
  const size_t cb = (size_t)b * 1024 * 2048 + (size_t)h * 128;
  for (int r = 0; r < 8; ++r) {
    const int qq = qbase + r + half * 8;
    const float inv = 1.0f / fmaxf(lr[r], 1e-9f);
    bf16* orow = ctx + cb + (size_t)qq * 2048;
    for (int nd = 0; nd < 8; ++nd)
      orow[nd * 16 + l16] = (bf16)(acc[nd][r] * inv);
  }
}

// ---------------------------------------------------------------------------
// K3: output projection, bf16 A from workspace, f32 out + bias,
// register double-buffered staging.
// ---------------------------------------------------------------------------
__global__ __launch_bounds__(256)
void k_proj(const bf16* __restrict__ A, const float* __restrict__ W,
            const float* __restrict__ bias, float* __restrict__ out) {
  const int K = 2048, N = 2048;
  __shared__ bf16 lA[128 * 40];
  __shared__ bf16 lB[128 * 40];
  const int tid  = threadIdx.x;
  const int lane = tid & 31, wave = tid >> 5;
  const int half = lane >> 4, l16 = lane & 15;
  const int kb = half * 8, kh = half * 16;
  const int m0 = blockIdx.x * 128;
  const int n0 = blockIdx.y * 128;
  const int wm = (wave & 3) * 32;
  const int wn = (wave >> 2) * 64;

  v8f acc[2][4];
  for (int i = 0; i < 2; ++i)
    for (int j = 0; j < 4; ++j) acc[i][j] = vzero8();

  v8bf  ga[2];
  float4 gb[4];
  auto load_regs = [&](int k0) {
    for (int u = 0; u < 2; ++u) {
      const int ia = tid + u * 256;
      ga[u] = *(const v8bf*)(A + (size_t)(m0 + (ia >> 2)) * K + k0 + ((ia & 3) << 3));
    }
    for (int u = 0; u < 4; ++u) {
      const int ia = tid + u * 256;
      gb[u] = *(const float4*)(W + (size_t)(k0 + (ia >> 5)) * N + n0 + ((ia & 31) << 2));
    }
  };

  load_regs(0);
  for (int k0 = 0; k0 < K; k0 += 32) {
    __syncthreads();
    for (int u = 0; u < 2; ++u) {
      const int ia = tid + u * 256;
      *(v8bf*)&lA[(ia >> 2) * 40 + ((ia & 3) << 3)] = ga[u];
    }
    for (int u = 0; u < 4; ++u) {
      const int ia = tid + u * 256;
      const int r = ia >> 5, c4 = (ia & 31) << 2;
      lB[(c4 + 0) * 40 + r] = (bf16)gb[u].x;
      lB[(c4 + 1) * 40 + r] = (bf16)gb[u].y;
      lB[(c4 + 2) * 40 + r] = (bf16)gb[u].z;
      lB[(c4 + 3) * 40 + r] = (bf16)gb[u].w;
    }
    __syncthreads();
    if (k0 + 32 < K) load_regs(k0 + 32);
    v16bf af[2], bfr[4];
    for (int i = 0; i < 2; ++i) af[i] = fragA(&lA[(wm + i * 16 + l16) * 40], kb);
    for (int j = 0; j < 4; ++j) bfr[j] = fragB(&lB[(wn + j * 16 + l16) * 40], kh);
    for (int i = 0; i < 2; ++i)
      for (int j = 0; j < 4; ++j)
        acc[i][j] = WMMA_B16(af[i], bfr[j], acc[i][j]);
  }

  for (int i = 0; i < 2; ++i)
    for (int j = 0; j < 4; ++j) {
      const int col = n0 + wn + j * 16 + l16;
      const float bv = bias[col];
      for (int r = 0; r < 8; ++r) {
        const int row = m0 + wm + i * 16 + r + half * 8;
        out[(size_t)row * N + col] = acc[i][j][r] + bv;
      }
    }
}

// ---------------------------------------------------------------------------
extern "C" void kernel_launch(void* const* d_in, const int* in_sizes, int n_in,
                              void* d_out, int out_size, void* d_ws, size_t ws_size,
                              hipStream_t stream) {
  (void)in_sizes; (void)n_in; (void)out_size; (void)ws_size;
  const float* hidden    = (const float*)d_in[0];
  const float* W_attn    = (const float*)d_in[1];
  const float* b_attn    = (const float*)d_in[2];
  const float* W_proj    = (const float*)d_in[3];
  const float* b_proj    = (const float*)d_in[4];
  const float* k_pages   = (const float*)d_in[5];
  const float* v_pages   = (const float*)d_in[6];
  const int*   page_pos  = (const int*)d_in[7];
  const int*   input_pos = (const int*)d_in[8];
  float* out = (float*)d_out;

  // workspace: bf16 buffers. SZ = B*H*Q*Dh = 16.78M elems.
  // [qws SZ][kws SZ][vws_t SZ][ctx SZ][kcat 2*SZ][vt 2*SZ] = 8*SZ elems = 268MB.
  const size_t SZ = (size_t)8 * 16 * 1024 * 128;
  bf16* qws   = (bf16*)d_ws;
  bf16* kws   = qws + SZ;
  bf16* vws_t = kws + SZ;
  bf16* ctx   = vws_t + SZ;
  bf16* kcat  = ctx + SZ;
  bf16* vt    = kcat + 2 * SZ;

  const int attn_lds = 90112;  // 2*17408(K) + 2*18432(V) + 18432(P) bytes
  (void)hipFuncSetAttribute((const void*)k_attn,
                            hipFuncAttributeMaxDynamicSharedMemorySize, attn_lds);

  dim3 blk(256);
  k_kvconv<<<dim3(8, 8, 16), blk, 0, stream>>>(k_pages, v_pages, kcat, vt);
  k_qkv   <<<dim3(64, 48),   blk, 0, stream>>>(hidden, W_attn, b_attn, qws, kws, vws_t);
  k_attn  <<<dim3(8, 16, 8), blk, attn_lds, stream>>>(qws, kws, vws_t, kcat, vt,
                                                      page_pos, input_pos, ctx);
  k_proj  <<<dim3(64, 16),   blk, 0, stream>>>(ctx, W_proj, b_proj, out);
}
